// Attention_12532714569787
// MI455X (gfx1250) — matrix operand order repaired and checked by reading, not compiled, not run on hardware
//
#include <hip/hip_runtime.h>
#include <hip/hip_fp16.h>

// Problem sizes (fixed by the reference).
#define BQ  8
#define LQN 2048
#define LKN 2048
#define DN  1024

typedef __attribute__((ext_vector_type(16))) _Float16 v16h;
typedef __attribute__((ext_vector_type(8)))  _Float16 v8h;
typedef __attribute__((ext_vector_type(4)))  _Float16 v4h;
typedef __attribute__((ext_vector_type(8)))  float    v8f;
typedef __attribute__((ext_vector_type(4)))  float    v4f;
typedef __attribute__((ext_vector_type(4)))  int      v4i;

#define AS1 __attribute__((address_space(1)))
#define AS3 __attribute__((address_space(3)))

// ---- gfx1250 async global->LDS path (guarded so either way compiles) ----
#if __has_builtin(__builtin_amdgcn_global_load_async_to_lds_b128)
#define HAVE_ASYNC_LDS 1
#else
#define HAVE_ASYNC_LDS 0
#endif

__device__ __forceinline__ void async_g2l_b128(const void* g, void* l) {
#if HAVE_ASYNC_LDS
  // Probe-confirmed: param0 is addrspace(1) int4* ("__device__"), 4 args total.
  __builtin_amdgcn_global_load_async_to_lds_b128((AS1 v4i*)(void*)g, (AS3 v4i*)l, 0, 0);
#endif
}

__device__ __forceinline__ void wait_async0() {
#if __has_builtin(__builtin_amdgcn_s_wait_asynccnt)
  __builtin_amdgcn_s_wait_asynccnt(0);
#else
  asm volatile("s_wait_asynccnt 0x0" ::: "memory");
#endif
}

// ---------------------------------------------------------------------------
// f32 -> f16 conversion (weights)
// ---------------------------------------------------------------------------
__global__ __launch_bounds__(256)
void cvt_f32_f16_kernel(const float* __restrict__ src, _Float16* __restrict__ dst, int n4) {
  int i = blockIdx.x * 256 + threadIdx.x;
  if (i < n4) {
    v4f x = ((const v4f*)src)[i];
    v4h y;
    y.x = (_Float16)x.x; y.y = (_Float16)x.y; y.z = (_Float16)x.z; y.w = (_Float16)x.w;
    ((v4h*)dst)[i] = y;
  }
}

// ---------------------------------------------------------------------------
// Generic WMMA GEMM: C[m,n] = scale * sum_k A[m,k]*B'[k,n] (+ bias[n])
//   A_F16  : A elements are f16 (else f32, converted during LDS staging)
//   B_KN   : B source is row-major [K][N] (transpose-staged); else [N][K]
//   BIAS   : add f32 bias[n]
//   OUT_F16: write f16 (else f32)
// Block tile 128x128, BK=32, 256 threads = 8 waves, each wave 32(M)x64(N).
// Double-buffered LDS; f16 operands staged with GLOBAL_LOAD_ASYNC_TO_LDS_B128
// (ASYNCcnt) when available, f32/transpose operands via register prefetch.
// M,N multiples of 128; K multiple of 32 (true for all call sites).
// ---------------------------------------------------------------------------
template<bool A_F16, bool B_KN, bool BIAS, bool OUT_F16>
__global__ __launch_bounds__(256)
void gemm_wmma_kernel(const void* __restrict__ Av, const _Float16* __restrict__ Bmat,
                      const float* __restrict__ bias, void* __restrict__ Cv,
                      int M, int N, int K, long sA, long sB, long sC, float scale) {
  // 40-half row pitch (80B, 16B aligned) to break LDS bank conflicts.
  __shared__ _Float16 Asm[2][128][40];
  __shared__ _Float16 Bsm[2][128][40];

  const int t    = threadIdx.x;
  const int lane = t & 31;
  const int wid  = t >> 5;
  const int wm   = (wid & 3) * 32;   // wave M origin within block tile
  const int wn   = (wid >> 2) * 64;  // wave N origin within block tile
  const int m0   = blockIdx.y * 128;
  const int n0   = blockIdx.x * 128;
  const int z    = blockIdx.z;

  const float*    Af = (const float*)Av    + (size_t)z * sA;
  const _Float16* Ah = (const _Float16*)Av + (size_t)z * sA;
  const _Float16* Bp = Bmat                + (size_t)z * sB;

  v8f acc[2][4];
  const v8f vzero = {0.f,0.f,0.f,0.f,0.f,0.f,0.f,0.f};
  #pragma unroll
  for (int i = 0; i < 2; i++)
    #pragma unroll
    for (int j = 0; j < 4; j++) acc[i][j] = vzero;

  v4f prefA[4];   // register prefetch when A is f32
  v8h prefB[2];   // register prefetch when B needs transpose staging

  // ---- staging: issue (global side) / commit (LDS side for reg paths) ----
  auto issueA = [&](int k0, int bufi) {
    if constexpr (A_F16) {
      const int c8 = t & 3, r = t >> 2;
      #pragma unroll
      for (int rr = 0; rr < 128; rr += 64) {
        const _Float16* g = Ah + (size_t)(m0 + r + rr) * K + k0 + c8 * 8;
        _Float16* l = &Asm[bufi][r + rr][c8 * 8];
#if HAVE_ASYNC_LDS
        async_g2l_b128(g, l);
#else
        *(v8h*)l = *(const v8h*)g;
#endif
      }
    } else {
      const int c4 = t & 7, r = t >> 3;
      #pragma unroll
      for (int i = 0; i < 4; i++)
        prefA[i] = *(const v4f*)(Af + (size_t)(m0 + r + i * 32) * K + k0 + c4 * 4);
    }
  };
  auto commitA = [&](int bufi) {
    if constexpr (!A_F16) {
      const int c4 = t & 7, r = t >> 3;
      #pragma unroll
      for (int i = 0; i < 4; i++) {
        v4h y;
        y.x = (_Float16)prefA[i].x; y.y = (_Float16)prefA[i].y;
        y.z = (_Float16)prefA[i].z; y.w = (_Float16)prefA[i].w;
        *(v4h*)(&Asm[bufi][r + i * 32][c4 * 4]) = y;
      }
    }
  };
  auto issueB = [&](int k0, int bufi) {
    if constexpr (!B_KN) {
      const int c8 = t & 3, r = t >> 2;
      #pragma unroll
      for (int rr = 0; rr < 128; rr += 64) {
        const _Float16* g = Bp + (size_t)(n0 + r + rr) * K + k0 + c8 * 8;
        _Float16* l = &Bsm[bufi][r + rr][c8 * 8];
#if HAVE_ASYNC_LDS
        async_g2l_b128(g, l);
#else
        *(v8h*)l = *(const v8h*)g;
#endif
      }
    } else {
      const int c8 = t & 15, kr = t >> 4;
      #pragma unroll
      for (int i = 0; i < 2; i++)
        prefB[i] = *(const v8h*)(Bp + (size_t)(k0 + kr + i * 16) * N + n0 + c8 * 8);
    }
  };
  auto commitB = [&](int bufi) {
    if constexpr (B_KN) {
      const int c8 = t & 15, kr = t >> 4;
      #pragma unroll
      for (int i = 0; i < 2; i++)
        #pragma unroll
        for (int j = 0; j < 8; j++) Bsm[bufi][c8 * 8 + j][kr + i * 16] = prefB[i][j];
    }
  };

  // ---- prologue: stage tile 0 into buffer 0 ----
  issueA(0, 0); issueB(0, 0);
  commitA(0);   commitB(0);
#if HAVE_ASYNC_LDS
  wait_async0();
#endif
  __syncthreads();

  int buf = 0;
  for (int k0 = 0; k0 < K; k0 += 32) {
    const int nk = k0 + 32;
    const bool has_next = nk < K;
    if (has_next) { issueA(nk, buf ^ 1); issueB(nk, buf ^ 1); }

    // ---- build fragments per ISA VGPR layouts, then WMMA ----
    {
      const int row = lane & 15, hsel = lane >> 4;
      // A 16x32 f16: lanes<16 hold K {0..7,16..23}, lanes>=16 hold K {8..15,24..31}
      v16h afrag[2];
      #pragma unroll
      for (int mt = 0; mt < 2; mt++) {
        const _Float16* ap = &Asm[buf][wm + mt * 16 + row][hsel * 8];
        v8h lo = *(const v8h*)ap;
        v8h hi = *(const v8h*)(ap + 16);
        #pragma unroll
        for (int j = 0; j < 8; j++) { afrag[mt][j] = lo[j]; afrag[mt][8 + j] = hi[j]; }
      }
      // B 32x16 f16: lanes<16 = col n, K {0..7,16..23}; lanes>=16 = col n, K {8..15,24..31}
      v16h bfrag[4];
      #pragma unroll
      for (int nt = 0; nt < 4; nt++) {
        const _Float16* bp = &Bsm[buf][wn + nt * 16 + row][hsel * 8];
        v8h lo = *(const v8h*)bp;
        v8h hi = *(const v8h*)(bp + 16);
        #pragma unroll
        for (int j = 0; j < 8; j++) { bfrag[nt][j] = lo[j]; bfrag[nt][8 + j] = hi[j]; }
      }
      #pragma unroll
      for (int mt = 0; mt < 2; mt++)
        #pragma unroll
        for (int nt = 0; nt < 4; nt++)
          acc[mt][nt] = __builtin_amdgcn_wmma_f32_16x16x32_f16(
              false, afrag[mt], false, bfrag[nt], (short)0, acc[mt][nt], false, false);
    }

    if (has_next) { commitA(buf ^ 1); commitB(buf ^ 1); }
#if HAVE_ASYNC_LDS
    wait_async0();
#endif
    __syncthreads();
    buf ^= 1;
  }

  // ---- epilogue: C/D layout: VGPR r -> M = r + 8*(lane>>4), N = lane&15 ----
  const int col  = lane & 15;
  const int hsel = lane >> 4;
  #pragma unroll
  for (int mt = 0; mt < 2; mt++) {
    #pragma unroll
    for (int nt = 0; nt < 4; nt++) {
      const int gn = n0 + wn + nt * 16 + col;
      float bb = 0.f;
      if constexpr (BIAS) bb = bias[gn];
      #pragma unroll
      for (int r8 = 0; r8 < 8; r8++) {
        const int gm = m0 + wm + mt * 16 + hsel * 8 + r8;
        const float val = acc[mt][nt][r8] * scale + bb;
        if constexpr (OUT_F16)
          ((_Float16*)Cv)[(size_t)z * sC + (size_t)gm * N + gn] = (_Float16)val;
        else
          ((float*)Cv)[(size_t)z * sC + (size_t)gm * N + gn] = val;
      }
    }
  }
}

// ---------------------------------------------------------------------------
// In-place masked softmax over rows of length LKN.
// Matches reference: S already holds QK^T/sqrt(D); mask==0 -> -1e9; softmax.
// One 256-thread block per row; 8 elements/thread.
// ---------------------------------------------------------------------------
__global__ __launch_bounds__(256)
void softmax_mask_kernel(float* __restrict__ score, const int* __restrict__ mask) {
  const int rowid = blockIdx.x;        // 0 .. BQ*LQN-1
  const int b     = rowid >> 11;       // / LQN (2048)
  float* s = score + (size_t)rowid * LKN;
  const int* mrow = mask + (size_t)b * LKN;
  const int t = threadIdx.x;

  float vals[8];
  float vmax = -3.4e38f;
  #pragma unroll
  for (int i = 0; i < 8; i++) {
    const int k = t + i * 256;
    float x = s[k];
    x = (mrow[k] == 0) ? -1.0e9f : x;
    vals[i] = x;
    vmax = fmaxf(vmax, x);
  }
  __shared__ float red[256];
  red[t] = vmax; __syncthreads();
  #pragma unroll
  for (int off = 128; off > 0; off >>= 1) {
    if (t < off) red[t] = fmaxf(red[t], red[t + off]);
    __syncthreads();
  }
  const float rmax = red[0];
  __syncthreads();

  float sum = 0.f;
  #pragma unroll
  for (int i = 0; i < 8; i++) { vals[i] = __expf(vals[i] - rmax); sum += vals[i]; }
  red[t] = sum; __syncthreads();
  #pragma unroll
  for (int off = 128; off > 0; off >>= 1) {
    if (t < off) red[t] += red[t + off];
    __syncthreads();
  }
  const float inv = 1.0f / red[0];   // >= 1 always (max element -> exp(0)=1)
  #pragma unroll
  for (int i = 0; i < 8; i++) s[t + i * 256] = vals[i] * inv;
}

// ---------------------------------------------------------------------------
extern "C" void kernel_launch(void* const* d_in, const int* in_sizes, int n_in,
                              void* d_out, int out_size, void* d_ws, size_t ws_size,
                              hipStream_t stream) {
  const float* key   = (const float*)d_in[0];  // [B, LK, D]
  const float* query = (const float*)d_in[1];  // [B, LQ, D]
  const int*   mask  = (const int*)d_in[2];    // [B, LK]
  const float* Wq    = (const float*)d_in[3];
  const float* bq    = (const float*)d_in[4];
  const float* Wk    = (const float*)d_in[5];
  const float* bk    = (const float*)d_in[6];
  const float* Wv    = (const float*)d_in[7];
  const float* bv    = (const float*)d_in[8];

  float* outO = (float*)d_out;                          // [B, LQ, D]
  float* outS = outO + (size_t)BQ * LQN * DN;           // [B, LQ, LK]

  // Workspace layout (f16): Wq|Wk|Wv (1M each), Qh|Kh|Vh (16M each) = 102 MB.
  _Float16* ws  = (_Float16*)d_ws;
  const size_t WSZ = (size_t)DN * DN;                   // 1,048,576
  _Float16* Wqh = ws;
  _Float16* Wkh = ws + WSZ;
  _Float16* Wvh = ws + 2 * WSZ;
  _Float16* Qh  = ws + 3 * WSZ;
  _Float16* Kh  = Qh + (size_t)BQ * LQN * DN;
  _Float16* Vh  = Kh + (size_t)BQ * LKN * DN;

  // 1) weights -> f16
  const int wn4 = (int)(WSZ / 4);
  cvt_f32_f16_kernel<<<wn4 / 256, 256, 0, stream>>>(Wq, Wqh, wn4);
  cvt_f32_f16_kernel<<<wn4 / 256, 256, 0, stream>>>(Wk, Wkh, wn4);
  cvt_f32_f16_kernel<<<wn4 / 256, 256, 0, stream>>>(Wv, Wvh, wn4);

  const dim3 blk(256);

  // 2) projections: q = query@Wq^T+bq, k = key@Wk^T+bk (f32 A), v = k@Wv^T+bv (f16 A)
  {
    dim3 g(DN / 128, (BQ * LQN) / 128, 1);
    gemm_wmma_kernel<false, false, true, true><<<g, blk, 0, stream>>>(
        query, Wqh, bq, Qh, BQ * LQN, DN, DN, 0, 0, 0, 1.0f);
    gemm_wmma_kernel<false, false, true, true><<<g, blk, 0, stream>>>(
        key, Wkh, bk, Kh, BQ * LKN, DN, DN, 0, 0, 0, 1.0f);
    gemm_wmma_kernel<true, false, true, true><<<g, blk, 0, stream>>>(
        Kh, Wvh, bv, Vh, BQ * LKN, DN, DN, 0, 0, 0, 1.0f);
  }

  // 3) S = Q@K^T / sqrt(D) -> score region of d_out (f32)
  {
    dim3 g(LKN / 128, LQN / 128, BQ);
    gemm_wmma_kernel<true, false, false, false><<<g, blk, 0, stream>>>(
        Qh, Kh, nullptr, outS, LQN, LKN, DN,
        (long)LQN * DN, (long)LKN * DN, (long)LQN * LKN, 0.03125f);
  }

  // 4) mask + softmax, in place on d_out score region
  softmax_mask_kernel<<<BQ * LQN, 256, 0, stream>>>(outS, mask);

  // 5) O = P @ V (A = f32 probabilities from d_out, converted in staging; B = Vh [K][N])
  {
    dim3 g(DN / 128, LQN / 128, BQ);
    gemm_wmma_kernel<false, true, false, false><<<g, blk, 0, stream>>>(
        outS, Vh, nullptr, outO, LQN, DN, LKN,
        (long)LQN * LKN, (long)LKN * DN, (long)LQN * DN, 1.0f);
  }
}
